// Model_60954175865409
// MI455X (gfx1250) — matrix operand order repaired and checked by reading, not compiled, not run on hardware
//
#include <hip/hip_runtime.h>

#define DH    128
#define DIN_  64
#define NNODES 100000
#define NLEV  15

typedef __attribute__((ext_vector_type(16))) _Float16 v16h;
typedef __attribute__((ext_vector_type(8)))  float    v8f;

__device__ __constant__ int kCodes[5] = {3, 2, 5, 1, 4};  // AND, NOT, XOR, MAJ, OR

__device__ __forceinline__ v8f wmma16(v16h a, v16h b, v8f c) {
  // D = A(16x32 f16) * B(32x16 f16) + C(16x16 f32)
  return __builtin_amdgcn_wmma_f32_16x16x32_f16(false, a, false, b, (short)0, c,
                                                false, false);
}

// A-operand loader (16x32 f16 tile): lane L holds row M=L&15;
// halves 0..7 = K kb..kb+7, halves 8..15 = K kb+16..kb+23, kb = (L&16)?8:0.
__device__ __forceinline__ v16h a_from_f32(const float* __restrict__ row, int k0) {
  int kb = k0 + ((threadIdx.x & 16) ? 8 : 0);
  v16h a;
#pragma unroll
  for (int j = 0; j < 8; ++j) {
    a[j]     = (_Float16)row[kb + j];
    a[j + 8] = (_Float16)row[kb + 16 + j];
  }
  return a;
}

__device__ __forceinline__ v16h a_from_f16(const _Float16* __restrict__ row, int k0) {
  int kb = k0 + ((threadIdx.x & 16) ? 8 : 0);
  v16h a;
#pragma unroll
  for (int j = 0; j < 8; ++j) {
    a[j]     = row[kb + j];
    a[j + 8] = row[kb + 16 + j];
  }
  return a;
}

// B tiles are pre-packed: tile = 32 lanes x 16 halves, lane-major, contiguous.
__device__ __forceinline__ v16h b_tile(const _Float16* __restrict__ packed, int tile) {
  return *(const v16h*)(packed + (size_t)tile * 512 + (size_t)(threadIdx.x & 31) * 16);
}

// ---------------- weight packing: f32 KxNn row-major -> f16 WMMA B-tile layout
__global__ void pack_b_kernel(const float* __restrict__ src, _Float16* __restrict__ dst,
                              int K, int Nn) {
  int Ntl = Nn >> 4;
  int tiles = (K >> 5) * Ntl;
  int t = blockIdx.x * blockDim.x + threadIdx.x;
  if (t >= tiles * 32) return;
  int tile = t >> 5, lane = t & 31;
  int kt = tile / Ntl, nt = tile - kt * Ntl;
  const float* s = src + (size_t)blockIdx.z * K * Nn;
  _Float16* d = dst + (size_t)blockIdx.z * tiles * 512 + (size_t)tile * 512 + lane * 16;
  int n = nt * 16 + (lane & 15);
  int kbase = kt * 32 + ((lane & 16) ? 8 : 0);
#pragma unroll
  for (int j = 0; j < 16; ++j) d[j] = (_Float16)s[(size_t)(kbase + j + ((j >= 8) ? 8 : 0)) * Nn + n];
}

// ---------------- projection: s = x @ W_s, t = x @ W_t ; hf = 0
__global__ __launch_bounds__(256)
void proj_kernel(const float* __restrict__ x,
                 const _Float16* __restrict__ pWs, const _Float16* __restrict__ pWt,
                 float* __restrict__ s, float* __restrict__ t, float* __restrict__ hf) {
  int wave = threadIdx.x >> 5, lane = threadIdx.x & 31;
  int row0 = blockIdx.x * 16;
  const float* arow = x + (size_t)(row0 + (lane & 15)) * DIN_;
  int nt = wave;
  int n = nt * 16 + (lane & 15);
  int mhi = (lane & 16) ? 8 : 0;
  v8f accS = {}; v8f accT = {};
#pragma unroll
  for (int kt = 0; kt < 2; ++kt) {
    v16h a = a_from_f32(arow, kt * 32);
    accS = wmma16(a, b_tile(pWs, kt * 8 + nt), accS);
    accT = wmma16(a, b_tile(pWt, kt * 8 + nt), accT);
  }
#pragma unroll
  for (int v = 0; v < 8; ++v) {
    size_t r = (size_t)(row0 + v + mhi);
    s[r * DH + n] = accS[v];
    t[r * DH + n] = accT[v];
  }
#pragma unroll
  for (int i = 0; i < 8; ++i) {
    int idx = i * 256 + threadIdx.x;           // 0..2047 = 16 rows x 128
    hf[(size_t)row0 * DH + idx] = 0.f;
  }
}

__global__ void zero_kernel(float* __restrict__ p, size_t n) {
  size_t i = (size_t)blockIdx.x * blockDim.x + threadIdx.x;
  if (i < n) p[i] = 0.f;
}

// ---------------- deterministic per-type stream compaction (stable order).
// blockIdx.x == 0 -> edges (egate), blockIdx.x == 1 -> nodes (ngate).
__global__ __launch_bounds__(256)
void compact2_kernel(const int* __restrict__ eg, int ne, int pe,
                     int* __restrict__ el, int* __restrict__ ec,
                     const int* __restrict__ ng, int nn, int pn,
                     int* __restrict__ nl, int* __restrict__ nc) {
  const int* gates = blockIdx.x ? ng : eg;
  int n            = blockIdx.x ? nn : ne;
  int pitch        = blockIdx.x ? pn : pe;
  int* list        = blockIdx.x ? nl : el;
  int* cnt         = blockIdx.x ? nc : ec;

  __shared__ int base[5];
  __shared__ int wtot[8];
  if (threadIdx.x < 5) base[threadIdx.x] = 0;
  __syncthreads();
  int lane = threadIdx.x & 31, wv = threadIdx.x >> 5;
  for (int start = 0; start < n; start += 256) {
    int i = start + threadIdx.x;
    int g = (i < n) ? gates[i] : -1;
#pragma unroll
    for (int ti = 0; ti < 5; ++ti) {
      bool m = (g == kCodes[ti]);
      unsigned int bm = (unsigned int)__ballot(m);
      if (lane == 0) wtot[wv] = __popc(bm);
      __syncthreads();
      int pre = 0;
      for (int w = 0; w < wv; ++w) pre += wtot[w];
      int within = __popc(bm & ((1u << lane) - 1u));
      if (m) list[(size_t)ti * pitch + base[ti] + pre + within] = i;
      __syncthreads();
      if (threadIdx.x == 0) {
        int tot = 0;
        for (int w = 0; w < 8; ++w) tot += wtot[w];
        base[ti] += tot;
      }
      __syncthreads();
    }
  }
  if (threadIdx.x < 5) cnt[threadIdx.x] = base[threadIdx.x];
}

// ---------------- per-level, per-type 3-layer edge MLP + scatter-add
// Dense over compacted per-type edge list; the concat(s,hf) gather is fused
// into the K-tiling of the first matmul (K 0..127 = s row, 128..255 = hf row).
__global__ __launch_bounds__(256)
void edge_mlp_kernel(const float* __restrict__ s, const float* __restrict__ hf,
                     const int* __restrict__ src_l,
                     const _Float16* __restrict__ pW1, const _Float16* __restrict__ pW2,
                     const _Float16* __restrict__ pW3,
                     const float* __restrict__ b1, const float* __restrict__ b2,
                     const float* __restrict__ b3,
                     const int* __restrict__ elist, const int* __restrict__ ecnt,
                     const int* __restrict__ dpos_l,
                     float* __restrict__ msg, int pitchE, int NPR) {
  int ti = blockIdx.y;
  int e0 = blockIdx.x * 16;
  int cntE = ecnt[ti];
  if (e0 >= cntE) return;
  const int* el = elist + (size_t)ti * pitchE;

  __shared__ _Float16 lds1[16 * DH];
  __shared__ _Float16 lds2[16 * DH];

  int wave = threadIdx.x >> 5, lane = threadIdx.x & 31;
  int mrow = e0 + (lane & 15);
  int eid = el[(mrow < cntE) ? mrow : (cntE - 1)];
  int sn = src_l[eid];
  const float* srow = s + (size_t)sn * DH;
  const float* hrow = hf + (size_t)sn * DH;
  int nt = wave;
  int n = nt * 16 + (lane & 15);
  int mhi = (lane & 16) ? 8 : 0;
  const _Float16* W1p = pW1 + (size_t)ti * 64 * 512;   // 8 kt x 8 nt
  const _Float16* W2p = pW2 + (size_t)ti * 32 * 512;   // 4 kt x 8 nt
  const _Float16* W3p = pW3 + (size_t)ti * 32 * 512;
  __builtin_prefetch(W2p + (size_t)nt * 512, 0, 3);    // warm next stages (L2 hot)
  __builtin_prefetch(W3p + (size_t)nt * 512, 0, 3);

  { // h1 = relu(concat(s,hf)[src] @ W1 + b1)
    v8f acc = {};
#pragma unroll
    for (int kt = 0; kt < 4; ++kt)
      acc = wmma16(a_from_f32(srow, kt * 32), b_tile(W1p, kt * 8 + nt), acc);
#pragma unroll
    for (int kt = 4; kt < 8; ++kt)
      acc = wmma16(a_from_f32(hrow, (kt - 4) * 32), b_tile(W1p, kt * 8 + nt), acc);
    float bias = b1[ti * DH + n];
#pragma unroll
    for (int v = 0; v < 8; ++v) {
      float h = acc[v] + bias;
      lds1[(v + mhi) * DH + n] = (_Float16)(h > 0.f ? h : 0.f);
    }
  }
  __syncthreads();
  { // h2 = relu(h1 @ W2 + b2)
    const _Float16* a2 = lds1 + (size_t)(lane & 15) * DH;
    v8f acc = {};
#pragma unroll
    for (int kt = 0; kt < 4; ++kt)
      acc = wmma16(a_from_f16(a2, kt * 32), b_tile(W2p, kt * 8 + nt), acc);
    float bias = b2[ti * DH + n];
#pragma unroll
    for (int v = 0; v < 8; ++v) {
      float h = acc[v] + bias;
      lds2[(v + mhi) * DH + n] = (_Float16)(h > 0.f ? h : 0.f);
    }
  }
  __syncthreads();
  { // m = h2 @ W3 + b3, scatter-add into level slots
    const _Float16* a3 = lds2 + (size_t)(lane & 15) * DH;
    v8f acc = {};
#pragma unroll
    for (int kt = 0; kt < 4; ++kt)
      acc = wmma16(a_from_f16(a3, kt * 32), b_tile(W3p, kt * 8 + nt), acc);
    float bias = b3[ti * DH + n];
    float* mbase = msg + (size_t)ti * NPR * DH;
#pragma unroll
    for (int v = 0; v < 8; ++v) {
      int row = e0 + v + mhi;
      if (row < cntE) {
        int dp = dpos_l[el[row]];
        atomicAdd(&mbase[(size_t)dp * DH + n], acc[v] + bias);
      }
    }
  }
}

// ---------------- per-level, per-type GRU update (hf; and s for AND=code 3)
// dense over compacted per-type node list
__global__ __launch_bounds__(256)
void gru_kernel(const float* __restrict__ msg,
                const _Float16* __restrict__ pWih, const _Float16* __restrict__ pWhh,
                const float* __restrict__ bih, const float* __restrict__ bhh,
                const int* __restrict__ node_l,
                const int* __restrict__ nlist, const int* __restrict__ ncnt,
                float* __restrict__ s, float* __restrict__ hf, int pitchN, int NPR) {
  int ti = blockIdx.y;
  int code = kCodes[ti];
  int np0 = blockIdx.x * 16;
  int cntN = ncnt[ti];
  if (np0 >= cntN) return;
  const int* nl = nlist + (size_t)ti * pitchN;

  __shared__ float gi[16 * 384];
  __shared__ float gh[16 * 384];

  int wave = threadIdx.x >> 5, lane = threadIdx.x & 31;
  int row = np0 + (lane & 15);
  int orig = nl[(row < cntN) ? row : (cntN - 1)];   // original level slot
  int node = node_l[orig];
  const float* msgrow = msg + (size_t)ti * NPR * DH + (size_t)orig * DH;
  const float* hfrow  = hf + (size_t)node * DH;
  const _Float16* Wih = pWih + (size_t)ti * 96 * 512;  // 4 kt x 24 nt
  const _Float16* Whh = pWhh + (size_t)ti * 96 * 512;
  int mhi = (lane & 16) ? 8 : 0;
  __builtin_prefetch(Whh + (size_t)wave * 512, 0, 3);

  // gi = msg @ Wih + bih ; gh = hf_rows @ Whh + bhh   (24 N-tiles over 8 waves)
#pragma unroll
  for (int r3 = 0; r3 < 3; ++r3) {
    int nt = wave + 8 * r3;
    int n = nt * 16 + (lane & 15);
    v8f ai = {}; v8f ah = {};
#pragma unroll
    for (int kt = 0; kt < 4; ++kt) {
      ai = wmma16(a_from_f32(msgrow, kt * 32), b_tile(Wih, kt * 24 + nt), ai);
      ah = wmma16(a_from_f32(hfrow, kt * 32), b_tile(Whh, kt * 24 + nt), ah);
    }
    float bi = bih[ti * 384 + n], bh = bhh[ti * 384 + n];
#pragma unroll
    for (int v = 0; v < 8; ++v) {
      gi[(v + mhi) * 384 + n] = ai[v] + bi;
      gh[(v + mhi) * 384 + n] = ah[v] + bh;
    }
  }
  __syncthreads();

  // elementwise hf update
#pragma unroll
  for (int i = 0; i < 8; ++i) {
    int idx = i * 256 + threadIdx.x;           // 16 rows x 128 cols
    int mm = idx >> 7, dc = idx & 127;
    int rw = np0 + mm;
    if (rw < cntN) {
      int nd = node_l[nl[rw]];
      float ir = gi[mm * 384 + dc], iz = gi[mm * 384 + 128 + dc], in = gi[mm * 384 + 256 + dc];
      float hr = gh[mm * 384 + dc], hz = gh[mm * 384 + 128 + dc], hn = gh[mm * 384 + 256 + dc];
      float rg = 1.f / (1.f + __expf(-(ir + hr)));
      float zg = 1.f / (1.f + __expf(-(iz + hz)));
      float ng = tanhf(in + rg * hn);
      float hold = hf[(size_t)nd * DH + dc];
      hf[(size_t)nd * DH + dc] = (1.f - zg) * ng + zg * hold;
    }
  }

  if (code == 3) {  // AND gates also refresh s with the same gi, gh from s rows
    __syncthreads();
    const float* srow = s + (size_t)node * DH;
#pragma unroll
    for (int r3 = 0; r3 < 3; ++r3) {
      int nt = wave + 8 * r3;
      int n = nt * 16 + (lane & 15);
      v8f ah = {};
#pragma unroll
      for (int kt = 0; kt < 4; ++kt)
        ah = wmma16(a_from_f32(srow, kt * 32), b_tile(Whh, kt * 24 + nt), ah);
      float bh = bhh[ti * 384 + n];
#pragma unroll
      for (int v = 0; v < 8; ++v) gh[(v + mhi) * 384 + n] = ah[v] + bh;
    }
    __syncthreads();
#pragma unroll
    for (int i = 0; i < 8; ++i) {
      int idx = i * 256 + threadIdx.x;
      int mm = idx >> 7, dc = idx & 127;
      int rw = np0 + mm;
      if (rw < cntN) {
        int nd = node_l[nl[rw]];
        float ir = gi[mm * 384 + dc], iz = gi[mm * 384 + 128 + dc], in = gi[mm * 384 + 256 + dc];
        float hr = gh[mm * 384 + dc], hz = gh[mm * 384 + 128 + dc], hn = gh[mm * 384 + 256 + dc];
        float rg = 1.f / (1.f + __expf(-(ir + hr)));
        float zg = 1.f / (1.f + __expf(-(iz + hz)));
        float ng = tanhf(in + rg * hn);
        float sold = s[(size_t)nd * DH + dc];
        s[(size_t)nd * DH + dc] = (1.f - zg) * ng + zg * sold;
      }
    }
  }
}

extern "C" void kernel_launch(void* const* d_in, const int* in_sizes, int n_in,
                              void* d_out, int out_size, void* d_ws, size_t ws_size,
                              hipStream_t stream) {
  const float* x    = (const float*)d_in[0];
  const float* W_s  = (const float*)d_in[1];
  const float* W_t  = (const float*)d_in[2];
  const float* aW1  = (const float*)d_in[3];
  const float* ab1  = (const float*)d_in[4];
  const float* aW2  = (const float*)d_in[5];
  const float* ab2  = (const float*)d_in[6];
  const float* aW3  = (const float*)d_in[7];
  const float* ab3  = (const float*)d_in[8];
  const float* gWih = (const float*)d_in[9];
  const float* gWhh = (const float*)d_in[10];
  const float* gbih = (const float*)d_in[11];
  const float* gbhh = (const float*)d_in[12];
  const int* src_lv  = (const int*)d_in[13];
  const int* dpos_lv = (const int*)d_in[14];
  const int* egate_lv= (const int*)d_in[15];
  const int* node_lv = (const int*)d_in[16];
  const int* ngate_lv= (const int*)d_in[17];

  int Epad = in_sizes[13] / NLEV;
  int NP   = in_sizes[16] / NLEV;
  int EpadT = (Epad + 15) / 16, EpadR = EpadT * 16;
  int NPT   = (NP + 15) / 16,   NPR   = NPT * 16;

  float* s  = (float*)d_out;
  float* t  = s + (size_t)NNODES * DH;
  float* hf = t + (size_t)NNODES * DH;

  char* wp = (char*)d_ws;
  auto alloc = [&](size_t bytes) {
    char* p = wp;
    wp += (bytes + 63) & ~(size_t)63;
    return p;
  };
  _Float16* pWs  = (_Float16*)alloc((size_t)16 * 512 * 2);
  _Float16* pWt  = (_Float16*)alloc((size_t)16 * 512 * 2);
  _Float16* pW1  = (_Float16*)alloc((size_t)5 * 64 * 512 * 2);
  _Float16* pW2  = (_Float16*)alloc((size_t)5 * 32 * 512 * 2);
  _Float16* pW3  = (_Float16*)alloc((size_t)5 * 32 * 512 * 2);
  _Float16* pWih = (_Float16*)alloc((size_t)5 * 96 * 512 * 2);
  _Float16* pWhh = (_Float16*)alloc((size_t)5 * 96 * 512 * 2);
  float*    msg  = (float*)   alloc((size_t)5 * NPR * DH * 4);
  int*      elist= (int*)     alloc((size_t)5 * EpadR * 4);
  int*      nlist= (int*)     alloc((size_t)5 * NPR * 4);
  int*      ecnt = (int*)     alloc(8 * 4);
  int*      ncnt = (int*)     alloc(8 * 4);

  dim3 blk(256);
  auto packLaunch = [&](const float* src, _Float16* dst, int K, int Nn, int batch) {
    int thr = (K / 32) * (Nn / 16) * 32;
    dim3 grid((thr + 255) / 256, 1, batch);
    hipLaunchKernelGGL(pack_b_kernel, grid, blk, 0, stream, src, dst, K, Nn);
  };
  packLaunch(W_s,  pWs,  64,  128, 1);
  packLaunch(W_t,  pWt,  64,  128, 1);
  packLaunch(aW1,  pW1,  256, 128, 5);
  packLaunch(aW2,  pW2,  128, 128, 5);
  packLaunch(aW3,  pW3,  128, 128, 5);
  packLaunch(gWih, pWih, 128, 384, 5);
  packLaunch(gWhh, pWhh, 128, 384, 5);

  hipLaunchKernelGGL(proj_kernel, dim3(NNODES / 16), blk, 0, stream,
                     x, pWs, pWt, s, t, hf);

  size_t zn = (size_t)5 * NPR * DH;
  for (int l = 0; l < NLEV; ++l) {
    const int* src_l   = src_lv   + (size_t)l * Epad;
    const int* dpos_l  = dpos_lv  + (size_t)l * Epad;
    const int* egate_l = egate_lv + (size_t)l * Epad;
    const int* node_l  = node_lv  + (size_t)l * NP;
    const int* ngate_l = ngate_lv + (size_t)l * NP;

    hipLaunchKernelGGL(zero_kernel, dim3((unsigned)((zn + 255) / 256)), blk, 0, stream,
                       msg, zn);
    hipLaunchKernelGGL(compact2_kernel, dim3(2), blk, 0, stream,
                       egate_l, Epad, EpadR, elist, ecnt,
                       ngate_l, NP,   NPR,   nlist, ncnt);
    hipLaunchKernelGGL(edge_mlp_kernel, dim3(EpadT, 5), blk, 0, stream,
                       s, hf, src_l, pW1, pW2, pW3, ab1, ab2, ab3,
                       elist, ecnt, dpos_l, msg, EpadR, NPR);
    hipLaunchKernelGGL(gru_kernel, dim3(NPT, 5), blk, 0, stream,
                       msg, pWih, pWhh, gbih, gbhh, node_l,
                       nlist, ncnt, s, hf, NPR, NPR);
  }
}
